// DecoderTree_63385127355041
// MI455X (gfx1250) — matrix operand order repaired and checked by reading, not compiled, not run on hardware
//
#include <hip/hip_runtime.h>
#include <math.h>

// ---- problem constants (match reference) ----
#define Hh 512
#define Ee 128
#define NSTART 5
#define NCONST 2
#define NNUM 8
#define Bb 48
#define Ss 256
#define Tt 23
#define DEe 768
#define Oo (NCONST + NNUM)      // 10
#define GIN (2*Hh + Ee)         // 1152
#define DMAX 12                 // max merge chain depth (<= (T-1)/2 = 11)
#define NCAND (2*Tt + 2)        // 48 candidate goal slots
#define OUTW 15                 // NSTART + Oo

typedef __attribute__((ext_vector_type(16))) _Float16 v16h;
typedef __attribute__((ext_vector_type(8)))  _Float16 v8h;
typedef __attribute__((ext_vector_type(4)))  float    v4f;
typedef __attribute__((ext_vector_type(8)))  float    v8f;

// ---------------------------------------------------------------------
// Fragment loaders. Preconditions (guaranteed by host orchestration):
//   K % 32 == 0, M % 16 == 0, Bt is f16, column-major (Npad x K), Npad%16==0.
// A fragment (16x32 f16): lane row = m0+(lane&15); per-lane K runs are
// contiguous in row-major memory -> 4x float4 loads + cvt.
// B fragment (32x16 f16): lane col = n0+(lane&15); element i<8 -> K=k0+8g+i,
// i>=8 -> K=k0+16+8g+(i-8) in the transposed weight -> 2x 16B loads.
// ---------------------------------------------------------------------
__device__ __forceinline__ v16h load_a_frag(const float* __restrict__ ar /*row base + k0 + 8g*/) {
    v4f a0 = *(const v4f*)(ar);
    v4f a1 = *(const v4f*)(ar + 4);
    v4f a2 = *(const v4f*)(ar + 16);
    v4f a3 = *(const v4f*)(ar + 20);
    v16h af;
    af[0]=(_Float16)a0.x;  af[1]=(_Float16)a0.y;  af[2]=(_Float16)a0.z;  af[3]=(_Float16)a0.w;
    af[4]=(_Float16)a1.x;  af[5]=(_Float16)a1.y;  af[6]=(_Float16)a1.z;  af[7]=(_Float16)a1.w;
    af[8]=(_Float16)a2.x;  af[9]=(_Float16)a2.y;  af[10]=(_Float16)a2.z; af[11]=(_Float16)a2.w;
    af[12]=(_Float16)a3.x; af[13]=(_Float16)a3.y; af[14]=(_Float16)a3.z; af[15]=(_Float16)a3.w;
    return af;
}

__device__ __forceinline__ v16h load_b_frag(const _Float16* __restrict__ br /*col base + k0 + 8g*/) {
    v8h b0 = *(const v8h*)(br);
    v8h b1 = *(const v8h*)(br + 16);
    v16h bf;
#pragma unroll
    for (int i = 0; i < 8; ++i) { bf[i] = b0[i]; bf[i+8] = b1[i]; }
    return bf;
}

// =====================================================================
// Plain WMMA GEMM: C[M,N] = act(A[M,K] @ Bt^T + bias); act 0=none,1=tanh
// One wave32 per 16x16 C tile. Branch-free inner loop.
// =====================================================================
__global__ __launch_bounds__(32)
void gemm_wmma(const float* __restrict__ A, int lda,
               const _Float16* __restrict__ Bt,
               const float* __restrict__ bias,
               float* __restrict__ C, int ldc,
               int M, int N, int K, int act)
{
    const int lane = threadIdx.x & 31;
    const int g    = lane >> 4;
    const int lr   = lane & 15;
    const int m0   = blockIdx.y << 4;
    const int n0   = blockIdx.x << 4;
    const float*    arow = A  + (size_t)(m0 + lr) * lda + (g << 3);
    const _Float16* bcol = Bt + (size_t)(n0 + lr) * K   + (g << 3);

    v8f acc = {};
    for (int k0 = 0; k0 < K; k0 += 32) {
        __builtin_prefetch(arow + k0 + 32, 0, 1);
        __builtin_prefetch(bcol + k0 + 32, 0, 1);
        v16h af = load_a_frag(arow + k0);
        v16h bf = load_b_frag(bcol + k0);
        acc = __builtin_amdgcn_wmma_f32_16x16x32_f16(
                  false, af, false, bf, (short)0, acc, false, false);
    }
    const int col = n0 + lr;
    if (col < N) {
        float bv = bias ? bias[col] : 0.0f;
#pragma unroll
        for (int r = 0; r < 8; ++r) {
            int rr = m0 + r + (g << 3);
            float v = acc[r] + bv;
            if (act == 1) v = tanhf(v);
            C[(size_t)rr * ldc + col] = v;
        }
    }
}

// =====================================================================
// Fused gate GEMM: C = tanh(A@Bt1^T + b1) * sigmoid(A@Bt2^T + b2)
// Two WMMA accumulators per tile; optional per-row mask (merge chains):
//   mask==null -> always store; else store only rows with mask[row]!=0.
// =====================================================================
__global__ __launch_bounds__(32)
void gemm_gate(const float* __restrict__ A, int lda,
               const _Float16* __restrict__ Bt1,
               const _Float16* __restrict__ Bt2,
               const float* __restrict__ b1,
               const float* __restrict__ b2,
               float* __restrict__ C, int ldc,
               int M, int N, int K,
               const int* __restrict__ mask)
{
    const int lane = threadIdx.x & 31;
    const int g    = lane >> 4;
    const int lr   = lane & 15;
    const int m0   = blockIdx.y << 4;
    const int n0   = blockIdx.x << 4;
    const float*    arow  = A   + (size_t)(m0 + lr) * lda + (g << 3);
    const _Float16* bcol1 = Bt1 + (size_t)(n0 + lr) * K   + (g << 3);
    const _Float16* bcol2 = Bt2 + (size_t)(n0 + lr) * K   + (g << 3);

    v8f acc1 = {}, acc2 = {};
    for (int k0 = 0; k0 < K; k0 += 32) {
        __builtin_prefetch(arow + k0 + 32, 0, 1);
        v16h af  = load_a_frag(arow + k0);
        v16h bf1 = load_b_frag(bcol1 + k0);
        v16h bf2 = load_b_frag(bcol2 + k0);
        acc1 = __builtin_amdgcn_wmma_f32_16x16x32_f16(
                   false, af, false, bf1, (short)0, acc1, false, false);
        acc2 = __builtin_amdgcn_wmma_f32_16x16x32_f16(
                   false, af, false, bf2, (short)0, acc2, false, false);
    }
    const int col = n0 + lr;
    if (col < N) {
        float bb1 = b1[col], bb2 = b2[col];
#pragma unroll
        for (int r = 0; r < 8; ++r) {
            int rr = m0 + r + (g << 3);
            float v = tanhf(acc1[r] + bb1) * (1.0f / (1.0f + expf(-(acc2[r] + bb2))));
            if (!mask || mask[rr]) C[(size_t)rr * ldc + col] = v;
        }
    }
}

// =====================================================================
// Weight convert + transpose: W (K x N f32, row-major) -> Bt (Npad x K f16,
// column-major), zero-padded columns.
// =====================================================================
__global__ void convt_kernel(const float* __restrict__ W, _Float16* __restrict__ Bt,
                             int K, int N, int Npad) {
    int total = Npad * K;
    for (int i = blockIdx.x*blockDim.x + threadIdx.x; i < total; i += gridDim.x*blockDim.x) {
        int n = i / K, k = i % K;
        Bt[i] = (n < N) ? (_Float16)W[(size_t)k * N + n] : (_Float16)0.0f;
    }
}

// =====================================================================
// Plan kernel: one thread per batch item, mirrors _build_plan on y.
// =====================================================================
__global__ void plan_kernel(const int* __restrict__ y,
                            int* cur_idx, int* left_src, int* left_valid,
                            int* gen_input, int* num_idx, int* is_term,
                            int* m_op, int* m_tm, int* m_ac)
{
    int i = threadIdx.x;
    if (i >= Bb) return;
    int node[40]; int ntop = 0;
    int embs[40]; int etop = 0;   // (t<<1) | isTerm
    node[ntop++] = 1;
    int lcv = 0, lcs = 0;
    for (int t = 0; t < Tt; ++t) {
        left_valid[t*Bb + i] = lcv;
        left_src [t*Bb + i] = lcs;
        cur_idx  [t*Bb + i] = ntop ? node[ntop-1] : 0;
        int tok = y[i*Tt + t];
        gen_input[t*Bb + i] = (tok < NSTART) ? tok : 0;
        int ni = tok - NSTART; if (ni < 0) ni = 0;
        num_idx[t*Bb + i] = ni;
        is_term[t*Bb + i] = (tok >= NSTART) ? 1 : 0;
        for (int d = 0; d < DMAX; ++d) {
            m_op[(t*DMAX + d)*Bb + i] = 0;
            m_tm[(t*DMAX + d)*Bb + i] = 0;
            m_ac[(t*DMAX + d)*Bb + i] = 0;
        }
        if (ntop) {
            --ntop;
            if (tok < NSTART) {
                if (ntop < 38) { node[ntop++] = 3 + 2*t; node[ntop++] = 2 + 2*t; }
                if (etop < 40) embs[etop++] = (t << 1);            // op
            } else {
                int d = 0;
                while (etop >= 2 && (embs[etop-1] & 1) && d < DMAX) {
                    int term = embs[--etop];
                    int opn  = embs[--etop];
                    m_op[(t*DMAX + d)*Bb + i] = opn  >> 1;
                    m_tm[(t*DMAX + d)*Bb + i] = term >> 1;
                    m_ac[(t*DMAX + d)*Bb + i] = 1;
                    ++d;
                }
                if (etop < 40) embs[etop++] = (t << 1) | 1;        // term
            }
            if (etop > 0 && (embs[etop-1] & 1)) { lcv = 1; lcs = embs[etop-1] >> 1; }
            else                                { lcv = 0; lcs = 0; }
        } else {
            lcv = 0; lcs = 0;
        }
    }
}

// =====================================================================
// Elementwise / build kernels (grid-stride)
// =====================================================================
__global__ void fill0_kernel(float* p, int n) {
    for (int i = blockIdx.x*blockDim.x + threadIdx.x; i < n; i += gridDim.x*blockDim.x)
        p[i] = 0.0f;
}

__global__ void qsel_kernel(const float* gl, const float* gr, const int* lv,
                            float* q, int n) {
    for (int i = blockIdx.x*blockDim.x + threadIdx.x; i < n; i += gridDim.x*blockDim.x) {
        int b = i / Hh;
        q[i] = lv[b] ? gr[i] : gl[i];
    }
}

__global__ void build_cl_kernel(const float* cand, const float* terms,
                                const int* ci, const int* ls,
                                float* Xcl, float* Xcr, int n) {
    for (int i = blockIdx.x*blockDim.x + threadIdx.x; i < n; i += gridDim.x*blockDim.x) {
        int b = i / Hh, h = i % Hh;
        float c = cand [(size_t)ci[b]*Bb*Hh + i];
        float l = terms[(size_t)ls[b]*Bb*Hh + i];
        Xcl[i] = c;
        Xcr[(size_t)b*2*Hh + h]      = c;
        Xcr[(size_t)b*2*Hh + Hh + h] = l;
    }
}

__global__ void build_leaf_kernel(const float* q, const float* ctx, float* leaf, int n) {
    for (int i = blockIdx.x*blockDim.x + threadIdx.x; i < n; i += gridDim.x*blockDim.x) {
        int b = i / Hh, h = i % Hh;
        leaf[(size_t)b*2*Hh + h]      = q[i];
        leaf[(size_t)b*2*Hh + Hh + h] = ctx[i];
    }
}

__global__ void build_gin_kernel(const float* q, const float* ctx,
                                 const float* op_embed, const int* gi,
                                 float* labs_t, float* gin, int n) {
    for (int i = blockIdx.x*blockDim.x + threadIdx.x; i < n; i += gridDim.x*blockDim.x) {
        int b = i / GIN, c = i % GIN;
        float v;
        if (c < Hh)          v = q  [(size_t)b*Hh + c];
        else if (c < 2*Hh)   v = ctx[(size_t)b*Hh + (c - Hh)];
        else {
            int e = c - 2*Hh;
            v = op_embed[(size_t)gi[b]*Ee + e];
            labs_t[(size_t)b*Ee + e] = v;
        }
        gin[i] = v;
    }
}

__global__ void build_mi_kernel(const float* labs, const float* terms, const float* cur,
                                const int* op, const int* tm, float* mi, int n) {
    for (int i = blockIdx.x*blockDim.x + threadIdx.x; i < n; i += gridDim.x*blockDim.x) {
        int b = i / GIN, c = i % GIN;
        float v;
        if (c < Ee)            v = labs [(size_t)op[b]*Bb*Ee + (size_t)b*Ee + c];
        else if (c < Ee + Hh)  v = terms[(size_t)tm[b]*Bb*Hh + (size_t)b*Hh + (c - Ee)];
        else                   v = cur  [(size_t)b*Hh + (c - Ee - Hh)];
        mi[i] = v;
    }
}

__global__ void build_allnums_kernel(const float* emb, const float* const_embed,
                                     const int* xnp, const int* xnm, float* an, int n) {
    for (int i = blockIdx.x*blockDim.x + threadIdx.x; i < n; i += gridDim.x*blockDim.x) {
        int b = i / (Oo*Hh); int rem = i % (Oo*Hh);
        int o = rem / Hh, h = rem % Hh;
        float v;
        if (o < NCONST) v = const_embed[(size_t)o*Hh + h];
        else {
            int j = o - NCONST;
            int pos = xnp[b*NNUM + j];
            v = emb[((size_t)b*Ss + pos)*Hh + h] * (xnm[b*NNUM + j] ? 1.0f : 0.0f);
        }
        an[i] = v;
    }
}

__global__ void cur_init_kernel(const float* an, const int* nidx, float* cur, int n) {
    for (int i = blockIdx.x*blockDim.x + threadIdx.x; i < n; i += gridDim.x*blockDim.x) {
        int b = i / Hh, h = i % Hh;
        cur[i] = an[((size_t)b*Oo + nidx[b])*Hh + h];
    }
}

__global__ void terms_set_kernel(const float* cur, const int* ist, float* dst, int n) {
    for (int i = blockIdx.x*blockDim.x + threadIdx.x; i < n; i += gridDim.x*blockDim.x) {
        int b = i / Hh;
        dst[i] = ist[b] ? cur[i] : 0.0f;
    }
}

// =====================================================================
// Attention / score reduction kernels
// =====================================================================
__global__ __launch_bounds__(128)
void attn_score_kernel(const float* __restrict__ embW, const float* __restrict__ qW,
                       const float* __restrict__ v_attn, const float* __restrict__ bv,
                       const int* __restrict__ x, float* __restrict__ sc) {
    __shared__ float red[128];
    int b = blockIdx.y, s = blockIdx.x, tid = threadIdx.x;
    const float* ew = embW + ((size_t)b*Ss + s)*Hh;
    const float* qw = qW + (size_t)b*Hh;
    float p = 0.0f;
    for (int h = tid; h < Hh; h += 128) p += v_attn[h] * tanhf(ew[h] + qw[h]);
    red[tid] = p; __syncthreads();
    for (int st = 64; st > 0; st >>= 1) { if (tid < st) red[tid] += red[tid+st]; __syncthreads(); }
    if (tid == 0)
        sc[(size_t)b*Ss + s] = (x[(size_t)b*Ss + s] == 0) ? -1e12f : (red[0] + bv[0]);
}

__global__ __launch_bounds__(Ss)
void softmax_kernel(const float* sc, float* aw) {
    __shared__ float red[Ss];
    int b = blockIdx.x, tid = threadIdx.x;
    float v = sc[(size_t)b*Ss + tid];
    red[tid] = v; __syncthreads();
    for (int st = Ss/2; st > 0; st >>= 1) { if (tid < st) red[tid] = fmaxf(red[tid], red[tid+st]); __syncthreads(); }
    float mx = red[0]; __syncthreads();
    float e = expf(v - mx);
    red[tid] = e; __syncthreads();
    for (int st = Ss/2; st > 0; st >>= 1) { if (tid < st) red[tid] += red[tid+st]; __syncthreads(); }
    aw[(size_t)b*Ss + tid] = e / red[0];
}

__global__ __launch_bounds__(256)
void ctx_kernel(const float* __restrict__ aw, const float* __restrict__ emb,
                float* __restrict__ ctx) {
    __shared__ float as[Ss];
    int b = blockIdx.x, tid = threadIdx.x;
    as[tid] = aw[(size_t)b*Ss + tid];
    __syncthreads();
    for (int h = tid; h < Hh; h += 256) {
        float acc = 0.0f;
        const float* eb = emb + (size_t)b*Ss*Hh + h;
        for (int s = 0; s < Ss; ++s) acc += as[s] * eb[(size_t)s*Hh];
        ctx[(size_t)b*Hh + h] = acc;
    }
}

__global__ __launch_bounds__(128)
void num_score_kernel(const float* __restrict__ leafW, const float* __restrict__ numsW,
                      const float* __restrict__ v_score, const float* __restrict__ bv,
                      const int* __restrict__ xnm, float* __restrict__ out, int t) {
    __shared__ float red[128];
    int b = blockIdx.y, o = blockIdx.x, tid = threadIdx.x;
    const float* lw = leafW + (size_t)b*Hh;
    const float* nw = numsW + ((size_t)b*Oo + o)*Hh;
    float p = 0.0f;
    for (int h = tid; h < Hh; h += 128) p += v_score[h] * tanhf(lw[h] + nw[h]);
    red[tid] = p; __syncthreads();
    for (int st = 64; st > 0; st >>= 1) { if (tid < st) red[tid] += red[tid+st]; __syncthreads(); }
    if (tid == 0) {
        float v = red[0] + bv[0];
        if (o >= NCONST && xnm[b*NNUM + (o - NCONST)] == 0) v = -1e12f;
        out[((size_t)b*Tt + t)*OUTW + NSTART + o] = v;
    }
}

// =====================================================================
// Host orchestration
// =====================================================================
#define GEMM(Aptr,lda,Btp,biasp,Cptr,ldc,M,N,K,act) \
    gemm_wmma<<<dim3(((N)+15)/16, ((M)+15)/16), 32, 0, stream>>>( \
        Aptr, lda, Btp, biasp, Cptr, ldc, M, N, K, act)

#define GATE(Aptr,lda,Bt1,Bt2,bb1,bb2,Cptr,M,N,K,maskp) \
    gemm_gate<<<dim3(((N)+15)/16, ((M)+15)/16), 32, 0, stream>>>( \
        Aptr, lda, Bt1, Bt2, bb1, bb2, Cptr, (N), M, N, K, maskp)

#define EW(n) <<<((n)+255)/256, 256, 0, stream>>>

extern "C" void kernel_launch(void* const* d_in, const int* in_sizes, int n_in,
                              void* d_out, int out_size, void* d_ws, size_t ws_size,
                              hipStream_t stream) {
    const int*   x   = (const int*)  d_in[0];
    const int*   y   = (const int*)  d_in[1];
    const int*   xnm = (const int*)  d_in[2];
    const int*   xnp = (const int*)  d_in[3];
    const float* EE  = (const float*)d_in[4];   // (B,S,DE)
    const float* ES  = (const float*)d_in[5];   // (B,DE)
    int p = 6;
    const float* W_ei  = (const float*)d_in[p++]; const float* b_ei  = (const float*)d_in[p++];
    const float* W_si  = (const float*)d_in[p++]; const float* b_si  = (const float*)d_in[p++];
    const float* c_emb = (const float*)d_in[p++]; // (NCONST,H)
    const float* o_emb = (const float*)d_in[p++]; // (NSTART,E)
    const float* W_ops = (const float*)d_in[p++]; const float* b_ops = (const float*)d_in[p++];
    const float* W_attn= (const float*)d_in[p++]; const float* b_attn= (const float*)d_in[p++];
    const float* v_attn= (const float*)d_in[p++]; const float* bv_attn=(const float*)d_in[p++];
    const float* W_scr = (const float*)d_in[p++]; const float* b_scr = (const float*)d_in[p++];
    const float* v_scr = (const float*)d_in[p++]; const float* bv_scr= (const float*)d_in[p++];
    const float* W_cl  = (const float*)d_in[p++]; const float* b_cl  = (const float*)d_in[p++];
    const float* W_clg = (const float*)d_in[p++]; const float* b_clg = (const float*)d_in[p++];
    const float* W_cr  = (const float*)d_in[p++]; const float* b_cr  = (const float*)d_in[p++];
    const float* W_crg = (const float*)d_in[p++]; const float* b_crg = (const float*)d_in[p++];
    const float* W_gl  = (const float*)d_in[p++]; const float* b_gl  = (const float*)d_in[p++];
    const float* W_glg = (const float*)d_in[p++]; const float* b_glg = (const float*)d_in[p++];
    const float* W_gr  = (const float*)d_in[p++]; const float* b_gr  = (const float*)d_in[p++];
    const float* W_grg = (const float*)d_in[p++]; const float* b_grg = (const float*)d_in[p++];
    const float* W_m   = (const float*)d_in[p++]; const float* b_m   = (const float*)d_in[p++];
    const float* W_mg  = (const float*)d_in[p++]; const float* b_mg  = (const float*)d_in[p++];

    float* outF = (float*)d_out;

    // ---- workspace layout (16B-aligned byte allocator) ----
    char* base = (char*)d_ws;
    size_t off = 0;
    auto alloc = [&](size_t bytes) -> void* {
        void* r = base + off;
        off = (off + bytes + 15) & ~(size_t)15;
        return r;
    };
    float* emb    = (float*)alloc(sizeof(float)*(size_t)Bb*Ss*Hh);
    float* embW   = (float*)alloc(sizeof(float)*(size_t)Bb*Ss*Hh);
    float* cand   = (float*)alloc(sizeof(float)*(size_t)NCAND*Bb*Hh);   // zeroed
    float* terms  = (float*)alloc(sizeof(float)*(size_t)Tt*Bb*Hh);      // zeroed
    float* labs   = (float*)alloc(sizeof(float)*(size_t)Tt*Bb*Ee);      // zeroed
    float* allnum = (float*)alloc(sizeof(float)*(size_t)Bb*Oo*Hh);
    float* numsW  = (float*)alloc(sizeof(float)*(size_t)Bb*Oo*Hh);
    float* Xcl    = (float*)alloc(sizeof(float)*(size_t)Bb*Hh);
    float* Xcr    = (float*)alloc(sizeof(float)*(size_t)Bb*2*Hh);
    float* glb    = (float*)alloc(sizeof(float)*(size_t)Bb*Hh);
    float* grb    = (float*)alloc(sizeof(float)*(size_t)Bb*Hh);
    float* qb     = (float*)alloc(sizeof(float)*(size_t)Bb*Hh);
    float* qW     = (float*)alloc(sizeof(float)*(size_t)Bb*Hh);
    float* sc     = (float*)alloc(sizeof(float)*(size_t)Bb*Ss);
    float* aw     = (float*)alloc(sizeof(float)*(size_t)Bb*Ss);
    float* ctx    = (float*)alloc(sizeof(float)*(size_t)Bb*Hh);
    float* leaf   = (float*)alloc(sizeof(float)*(size_t)Bb*2*Hh);
    float* leafW  = (float*)alloc(sizeof(float)*(size_t)Bb*Hh);
    float* gin    = (float*)alloc(sizeof(float)*(size_t)Bb*GIN);
    float* mi     = (float*)alloc(sizeof(float)*(size_t)Bb*GIN);
    float* cur    = (float*)alloc(sizeof(float)*(size_t)Bb*Hh);
    // f16 transposed weights (Npad x K each)
    _Float16* Bt_ei   = (_Float16*)alloc(sizeof(_Float16)*(size_t)Hh*DEe);
    _Float16* Bt_si   = (_Float16*)alloc(sizeof(_Float16)*(size_t)Hh*DEe);
    _Float16* Bt_atq  = (_Float16*)alloc(sizeof(_Float16)*(size_t)Hh*Hh);
    _Float16* Bt_ate  = (_Float16*)alloc(sizeof(_Float16)*(size_t)Hh*Hh);
    _Float16* Bt_scl  = (_Float16*)alloc(sizeof(_Float16)*(size_t)Hh*2*Hh);
    _Float16* Bt_scn  = (_Float16*)alloc(sizeof(_Float16)*(size_t)Hh*Hh);
    _Float16* Bt_ops  = (_Float16*)alloc(sizeof(_Float16)*(size_t)16*2*Hh);
    _Float16* Bt_cl   = (_Float16*)alloc(sizeof(_Float16)*(size_t)Hh*Hh);
    _Float16* Bt_clg  = (_Float16*)alloc(sizeof(_Float16)*(size_t)Hh*Hh);
    _Float16* Bt_cr   = (_Float16*)alloc(sizeof(_Float16)*(size_t)Hh*2*Hh);
    _Float16* Bt_crg  = (_Float16*)alloc(sizeof(_Float16)*(size_t)Hh*2*Hh);
    _Float16* Bt_gl   = (_Float16*)alloc(sizeof(_Float16)*(size_t)Hh*GIN);
    _Float16* Bt_glg  = (_Float16*)alloc(sizeof(_Float16)*(size_t)Hh*GIN);
    _Float16* Bt_gr   = (_Float16*)alloc(sizeof(_Float16)*(size_t)Hh*GIN);
    _Float16* Bt_grg  = (_Float16*)alloc(sizeof(_Float16)*(size_t)Hh*GIN);
    _Float16* Bt_m    = (_Float16*)alloc(sizeof(_Float16)*(size_t)Hh*GIN);
    _Float16* Bt_mg   = (_Float16*)alloc(sizeof(_Float16)*(size_t)Hh*GIN);
    // plan integers
    int* p_ci  = (int*)alloc(sizeof(int)*Tt*Bb);
    int* p_ls  = (int*)alloc(sizeof(int)*Tt*Bb);
    int* p_lv  = (int*)alloc(sizeof(int)*Tt*Bb);
    int* p_gi  = (int*)alloc(sizeof(int)*Tt*Bb);
    int* p_ni  = (int*)alloc(sizeof(int)*Tt*Bb);
    int* p_it  = (int*)alloc(sizeof(int)*Tt*Bb);
    int* p_mop = (int*)alloc(sizeof(int)*Tt*DMAX*Bb);
    int* p_mtm = (int*)alloc(sizeof(int)*Tt*DMAX*Bb);
    int* p_mac = (int*)alloc(sizeof(int)*Tt*DMAX*Bb);

    // ---- plan + zero persistent accumulator buffers ----
    plan_kernel<<<1, 64, 0, stream>>>(y, p_ci, p_ls, p_lv, p_gi, p_ni, p_it,
                                      p_mop, p_mtm, p_mac);
    fill0_kernel<<<1024, 256, 0, stream>>>(cand,  NCAND*Bb*Hh);
    fill0_kernel<<<1024, 256, 0, stream>>>(terms, Tt*Bb*Hh);
    fill0_kernel<<<256,  256, 0, stream>>>(labs,  Tt*Bb*Ee);

    // ---- weight convert + transpose to f16 (one-time, reused 23-276x) ----
    convt_kernel EW(Hh*DEe)  (W_ei,  Bt_ei,  DEe,  Hh, Hh);
    convt_kernel EW(Hh*DEe)  (W_si,  Bt_si,  DEe,  Hh, Hh);
    convt_kernel EW(Hh*Hh)   (W_attn,                Bt_atq, Hh,   Hh, Hh);
    convt_kernel EW(Hh*Hh)   (W_attn + (size_t)Hh*Hh,Bt_ate, Hh,   Hh, Hh);
    convt_kernel EW(Hh*2*Hh) (W_scr,                 Bt_scl, 2*Hh, Hh, Hh);
    convt_kernel EW(Hh*Hh)   (W_scr + (size_t)2*Hh*Hh,Bt_scn,Hh,   Hh, Hh);
    convt_kernel EW(16*2*Hh) (W_ops,  Bt_ops, 2*Hh, NSTART, 16);
    convt_kernel EW(Hh*Hh)   (W_cl,   Bt_cl,  Hh,   Hh, Hh);
    convt_kernel EW(Hh*Hh)   (W_clg,  Bt_clg, Hh,   Hh, Hh);
    convt_kernel EW(Hh*2*Hh) (W_cr,   Bt_cr,  2*Hh, Hh, Hh);
    convt_kernel EW(Hh*2*Hh) (W_crg,  Bt_crg, 2*Hh, Hh, Hh);
    convt_kernel EW(Hh*GIN)  (W_gl,   Bt_gl,  GIN,  Hh, Hh);
    convt_kernel EW(Hh*GIN)  (W_glg,  Bt_glg, GIN,  Hh, Hh);
    convt_kernel EW(Hh*GIN)  (W_gr,   Bt_gr,  GIN,  Hh, Hh);
    convt_kernel EW(Hh*GIN)  (W_grg,  Bt_grg, GIN,  Hh, Hh);
    convt_kernel EW(Hh*GIN)  (W_m,    Bt_m,   GIN,  Hh, Hh);
    convt_kernel EW(Hh*GIN)  (W_mg,   Bt_mg,  GIN,  Hh, Hh);

    // ---- precompute (step-invariant GEMMs) ----
    GEMM(EE, DEe, Bt_ei, b_ei, emb, Hh, Bb*Ss, Hh, DEe, 1);                    // emb
    GEMM(ES, DEe, Bt_si, b_si, cand + (size_t)1*Bb*Hh, Hh, Bb, Hh, DEe, 1);    // summary
    GEMM(emb, Hh, Bt_ate, b_attn, embW, Hh, Bb*Ss, Hh, Hh, 0);                 // emb@W_attn[H:]
    build_allnums_kernel EW(Bb*Oo*Hh) (emb, c_emb, xnp, xnm, allnum, Bb*Oo*Hh);
    GEMM(allnum, Hh, Bt_scn, b_scr, numsW, Hh, Bb*Oo, Hh, Hh, 0);              // nums@W_score[2H:]

    const int BH = Bb*Hh;
    for (int t = 0; t < Tt; ++t) {
        // --- gather goal stacks, left/right child gates, goal select ---
        build_cl_kernel EW(BH) (cand, terms, p_ci + t*Bb, p_ls + t*Bb, Xcl, Xcr, BH);
        GATE(Xcl, Hh,   Bt_cl, Bt_clg, b_cl, b_clg, glb, Bb, Hh, Hh,   nullptr);
        GATE(Xcr, 2*Hh, Bt_cr, Bt_crg, b_cr, b_crg, grb, Bb, Hh, 2*Hh, nullptr);
        qsel_kernel EW(BH) (glb, grb, p_lv + t*Bb, qb, BH);

        // --- attention: q projection + score/softmax/context ---
        GEMM(qb, Hh, Bt_atq, (const float*)nullptr, qW, Hh, Bb, Hh, Hh, 0);
        attn_score_kernel<<<dim3(Ss, Bb), 128, 0, stream>>>(embW, qW, v_attn, bv_attn, x, sc);
        softmax_kernel<<<Bb, Ss, 0, stream>>>(sc, aw);
        ctx_kernel<<<Bb, 256, 0, stream>>>(aw, emb, ctx);
        build_leaf_kernel EW(BH) (qb, ctx, leaf, BH);

        // --- op logits + number scores ---
        GEMM(leaf, 2*Hh, Bt_ops, b_ops, outF + (size_t)t*OUTW, Tt*OUTW, Bb, NSTART, 2*Hh, 0);
        GEMM(leaf, 2*Hh, Bt_scl, (const float*)nullptr, leafW, Hh, Bb, Hh, 2*Hh, 0);
        num_score_kernel<<<dim3(Oo, Bb), 128, 0, stream>>>(leafW, numsW, v_scr, bv_scr,
                                                           xnm, outF, t);

        // --- GenerateNode: two child goal gates ---
        build_gin_kernel EW(Bb*GIN) (qb, ctx, o_emb, p_gi + t*Bb,
                                     labs + (size_t)t*Bb*Ee, gin, Bb*GIN);
        GATE(gin, GIN, Bt_gl, Bt_glg, b_gl, b_glg, cand + (size_t)(2 + 2*t)*Bb*Hh,
             Bb, Hh, GIN, nullptr);
        GATE(gin, GIN, Bt_gr, Bt_grg, b_gr, b_grg, cand + (size_t)(3 + 2*t)*Bb*Hh,
             Bb, Hh, GIN, nullptr);

        // --- merge chains, batched by depth with static masks ---
        cur_init_kernel EW(BH) (allnum, p_ni + t*Bb, cur, BH);
        for (int d = 0; d < DMAX; ++d) {
            const int* mop = p_mop + (t*DMAX + d)*Bb;
            const int* mtm = p_mtm + (t*DMAX + d)*Bb;
            const int* mac = p_mac + (t*DMAX + d)*Bb;
            build_mi_kernel EW(Bb*GIN) (labs, terms, cur, mop, mtm, mi, Bb*GIN);
            GATE(mi, GIN, Bt_m, Bt_mg, b_m, b_mg, cur, Bb, Hh, GIN, mac);
        }
        terms_set_kernel EW(BH) (cur, p_it + t*Bb, terms + (size_t)t*Bb*Hh, BH);
    }
}
